// TMN_19636590477905
// MI455X (gfx1250) — compile-verified, matrix operand order checked by reading, not run on hardware
//
#include <hip/hip_runtime.h>
#include <hip/hip_bf16.h>
#include <math.h>

// ---------------- CDNA5 / gfx1250 WMMA plumbing ----------------
typedef __bf16 bf16_t;
typedef __attribute__((ext_vector_type(16))) __bf16 v16bf;
typedef __attribute__((ext_vector_type(8)))  __bf16 v8bf;
typedef __attribute__((ext_vector_type(8)))  float  v8f;
typedef unsigned int u32x4 __attribute__((ext_vector_type(4)));
typedef int          i32x4 __attribute__((ext_vector_type(4)));
typedef int          i32x8 __attribute__((ext_vector_type(8)));

#ifndef __has_builtin
#define __has_builtin(x) 0
#endif
#if defined(__AMDGCN__) && __has_builtin(__builtin_amdgcn_tensor_load_to_lds)
#define HAVE_TDM 1
#else
#define HAVE_TDM 0
#endif

#define HB   32    // batch
#define TT   512   // seq len
#define HH   256   // hidden
#define HHE  128   // H/2 embedding dim

__device__ __forceinline__ bf16_t f2b(float f) {
    unsigned u = __float_as_uint(f);
    unsigned r = (u >> 16) & 1u;
    u += 0x7fffu + r;               // round-to-nearest-even
    unsigned short s = (unsigned short)(u >> 16);
    return __builtin_bit_cast(bf16_t, s);
}
__device__ __forceinline__ float b2f(bf16_t b) {
    unsigned short s = __builtin_bit_cast(unsigned short, b);
    return __uint_as_float(((unsigned)s) << 16);
}
__device__ __forceinline__ float sigf(float x) { return 1.f / (1.f + __expf(-x)); }
__device__ __forceinline__ float leakyf(float x) { return x > 0.f ? x : 0.1f * x; }

__device__ __forceinline__ v8f wmma_bf16(v16bf a, v16bf b, v8f c) {
    // D = A(16x32) * B(32x16) + C, f32 accumulate
    return __builtin_amdgcn_wmma_f32_16x16x32_bf16(
        false, a, false, b, (short)0, c, false, false);
}

// A or B operand from row-major [row][k] memory (ld = leading dim, elems).
// ISA 16-bit layout: lane<16 -> row=row16+lane, K = {kt..kt+7, kt+16..kt+23};
// lanes 16-31 -> same rows, K = {kt+8..kt+15, kt+24..kt+31}.
__device__ __forceinline__ v16bf load_rm_frag(const bf16_t* base, int ld,
                                              int row16, int kt) {
    int lane = threadIdx.x & 31;
    int r  = row16 + (lane & 15);
    int k0 = kt + ((lane >> 4) ? 8 : 0);
    const bf16_t* p = base + (size_t)r * ld + k0;
    v8bf lo = *(const v8bf*)p;
    v8bf hi = *(const v8bf*)(p + 16);
    v16bf out;
#pragma unroll
    for (int i = 0; i < 8; ++i) { out[i] = lo[i]; out[i + 8] = hi[i]; }
    return out;
}

// B operand when memory holds the matrix K-major-in-rows: element[k][n] = base[k*ld + n]
__device__ __forceinline__ v16bf load_cm_frag(const bf16_t* base, int ld,
                                              int col16, int kt) {
    int lane = threadIdx.x & 31;
    int n  = col16 + (lane & 15);
    int k0 = kt + ((lane >> 4) ? 8 : 0);
    v16bf out;
#pragma unroll
    for (int i = 0; i < 8; ++i) {
        out[i]     = base[(size_t)(k0 + i) * ld + n];
        out[i + 8] = base[(size_t)(k0 + 16 + i) * ld + n];
    }
    return out;
}

__device__ __forceinline__ void wait_tensorcnt0() {
#if __has_builtin(__builtin_amdgcn_s_wait_tensorcnt)
    __builtin_amdgcn_s_wait_tensorcnt(0);
#else
    asm volatile("s_wait_tensorcnt 0x0" ::: "memory");
#endif
}

// ---------------- kernels ----------------

// f32 -> bf16 weight conversion
__global__ void cvt_kernel(const float* __restrict__ src, bf16_t* __restrict__ dst, int n) {
    int i = blockIdx.x * blockDim.x + threadIdx.x;
    if (i < n) dst[i] = f2b(src[i]);
}

// embedding MLP (2 -> 128, leaky 0.1) + grid mask;  one block per (b,t), 128 threads
__global__ void embed_kernel(const float* __restrict__ x, const float* __restrict__ Wm,
                             const float* __restrict__ bm,
                             bf16_t* __restrict__ emb, int* __restrict__ mask) {
    int bt = blockIdx.x;              // b*T + t
    int j  = threadIdx.x;             // 0..127
    const float* xp = x + (size_t)bt * 4;
    float v = xp[0] * Wm[j * 2] + xp[1] * Wm[j * 2 + 1] + bm[j];
    emb[(size_t)bt * HHE + j] = f2b(leakyf(v));
    if (j == 0) {
        // g = clip(gy*128 + gx - 258 + 1, 0, 16384); mask = (int)g != 0
        float g = xp[3] * 128.f + xp[2] - 257.f;
        g = fminf(fmaxf(g, 0.f), 16384.f);
        mask[bt] = ((int)g) != 0;
    }
}

// S[b,i,j] = ma[b,i,:] . mb[b,j,:], masked to -1e9. One wave per 16x16 tile.
__global__ void scores_kernel(const bf16_t* __restrict__ ea, const bf16_t* __restrict__ eb,
                              const int* __restrict__ mka, const int* __restrict__ mkb,
                              bf16_t* __restrict__ S) {
    int wid = blockIdx.x * (blockDim.x >> 5) + (threadIdx.x >> 5);
    int b  = wid >> 10;               // 32*32 tiles per batch
    int rt = (wid >> 5) & 31;
    int ct = wid & 31;
    const bf16_t* A  = ea + (size_t)b * TT * HHE;
    const bf16_t* Bm = eb + (size_t)b * TT * HHE;
    v8f acc = {};
#pragma unroll
    for (int kt = 0; kt < HHE; kt += 32) {
        v16bf af = load_rm_frag(A,  HHE, rt * 16, kt);
        v16bf bf = load_rm_frag(Bm, HHE, ct * 16, kt);   // mb row n = S column n
        acc = wmma_bf16(af, bf, acc);
    }
    int lane = threadIdx.x & 31;
    int col  = ct * 16 + (lane & 15);
    int okc  = mkb[b * TT + col];
#pragma unroll
    for (int i = 0; i < 8; ++i) {
        int row = rt * 16 + i + ((lane >> 4) ? 8 : 0);
        int ok  = okc && mka[b * TT + row];
        S[(size_t)b * TT * TT + (size_t)row * TT + col] = f2b(ok ? acc[i] : -1e9f);
    }
}

// masked row softmax, one 256-thread block per (b,row)
__global__ void softmax_kernel(const bf16_t* __restrict__ S, const int* __restrict__ mka,
                               const int* __restrict__ mkb, bf16_t* __restrict__ P) {
    __shared__ float red[256];
    int b = blockIdx.x >> 9, row = blockIdx.x & 511;
    const bf16_t* sr = S + (size_t)b * TT * TT + (size_t)row * TT;
    bf16_t*       pr = P + (size_t)b * TT * TT + (size_t)row * TT;
    int t = threadIdx.x;
    float s0 = b2f(sr[t]), s1 = b2f(sr[t + 256]);
    red[t] = fmaxf(s0, s1);
    __syncthreads();
    for (int o = 128; o > 0; o >>= 1) { if (t < o) red[t] = fmaxf(red[t], red[t + o]); __syncthreads(); }
    float m = red[0];
    __syncthreads();
    int v0 = mkb[b * TT + t], v1 = mkb[b * TT + t + 256];
    float e0 = v0 ? __expf(s0 - m) : 0.f;
    float e1 = v1 ? __expf(s1 - m) : 0.f;
    red[t] = e0 + e1;
    __syncthreads();
    for (int o = 128; o > 0; o >>= 1) { if (t < o) red[t] += red[t + o]; __syncthreads(); }
    float sum = red[0];
    float inv = (mka[b * TT + row] && sum > 0.f) ? 1.f / sum : 0.f;
    pr[t]       = f2b(e0 * inv);
    pr[t + 256] = f2b(e1 * inv);
}

// sum[b,i,h] = sum_j P[b,i,j] * eb[b,j,h]; one wave per 16x16 tile
__global__ void attnsum_kernel(const bf16_t* __restrict__ P, const bf16_t* __restrict__ eb,
                               bf16_t* __restrict__ sum) {
    int wid = blockIdx.x * (blockDim.x >> 5) + (threadIdx.x >> 5);
    int b  = wid >> 8;                // 32*8 tiles per batch
    int rt = (wid >> 3) & 31;
    int nt = wid & 7;
    const bf16_t* A  = P  + (size_t)b * TT * TT;
    const bf16_t* Bm = eb + (size_t)b * TT * HHE;
    v8f acc = {};
    for (int kt = 0; kt < TT; kt += 32) {
        v16bf af = load_rm_frag(A, TT, rt * 16, kt);
        v16bf bf = load_cm_frag(Bm, HHE, nt * 16, kt);   // eb is (j,h): K-major rows
        acc = wmma_bf16(af, bf, acc);
    }
    int lane = threadIdx.x & 31;
    int col  = nt * 16 + (lane & 15);
#pragma unroll
    for (int i = 0; i < 8; ++i) {
        int row = rt * 16 + i + ((lane >> 4) ? 8 : 0);
        sum[(size_t)b * TT * HHE + (size_t)row * HHE + col] = f2b(acc[i]);
    }
}

// X[t, b, 0:128] = ma[b,t,:]; X[t, b, 128:256] = ma - sum; one block per (b,t)
__global__ void pack_kernel(const bf16_t* __restrict__ e, const bf16_t* __restrict__ sum,
                            bf16_t* __restrict__ X) {
    int bt = blockIdx.x;
    int b = bt >> 9, t = bt & 511;
    int j = threadIdx.x;
    float m = b2f(e[(size_t)bt * HHE + j]);
    float s = b2f(sum[(size_t)bt * HHE + j]);
    size_t o = ((size_t)t * HB + b) * HH;
    X[o + j]       = f2b(m);
    X[o + HHE + j] = f2b(m - s);
}

// G = X @ Wih^T + (bih+bhh); M = T*B rows, N = 1024; one wave per 16x16 tile.
// Each wave stages its 16x256 bf16 Wih panel into LDS with the Tensor Data Mover
// (tensor_load_to_lds, 2-D descriptor), waits on TENSORcnt, then runs the WMMA
// k-loop with the B operand fed from LDS.
__global__ void ggemm_kernel(const bf16_t* __restrict__ X, const bf16_t* __restrict__ Wihb,
                             const float* __restrict__ bih, const float* __restrict__ bhh,
                             bf16_t* __restrict__ G) {
    int wv  = threadIdx.x >> 5;
    int wid = blockIdx.x * (blockDim.x >> 5) + wv;
    int mt = wid >> 6;                // 1024 M-tiles
    int nt = wid & 63;                // 64 N-tiles
#if HAVE_TDM
    extern __shared__ char gsm[];
    bf16_t* wst = ((bf16_t*)gsm) + (size_t)wv * (16 * 256);   // 8 KB panel per wave
    {
        int snt = __builtin_amdgcn_readfirstlane(nt);
        unsigned lds_off = (unsigned)(size_t)(void*)wst;       // flat addr low 32 = LDS offset
        unsigned long long ga =
            (unsigned long long)(const void*)(Wihb + (size_t)snt * 16 * 256);
        // D# group0: count=1 | lds_addr[63:32] | global_addr[120:64] | type=2
        u32x4 g0 = { 1u, lds_off, (unsigned)ga, (unsigned)(ga >> 32) | 0x80000000u };
        // D# group1: data_size=1(2B) @[17:16]; tensor_dim0=256 @[79:48];
        //            tensor_dim1=16 @[111:80]; tile_dim0=256 @[127:112];
        //            tile_dim1=16 @[143:128]; dim0_stride=256 @[207:160];
        //            dim1_stride=4096 @[255:208]
        i32x8 g1 = { 0x00010000, 0x01000000, 0x00100000, 0x01000000,
                     16, 256, 0x10000000, 0 };
        i32x4 z4 = { 0, 0, 0, 0 };
#if __clang_major__ >= 23
        i32x8 z8 = { 0, 0, 0, 0, 0, 0, 0, 0 };
        __builtin_amdgcn_tensor_load_to_lds(g0, g1, z4, z4, z8, 0);
#else
        __builtin_amdgcn_tensor_load_to_lds(g0, g1, z4, z4, 0);
#endif
        wait_tensorcnt0();   // per-wave: panel is private to this wave
    }
    const bf16_t* Bsrc = wst;
    const int brow = 0;
#else
    const bf16_t* Bsrc = Wihb;
    const int brow = nt * 16;
#endif
    v8f acc = {};
#pragma unroll
    for (int kt = 0; kt < HH; kt += 32) {
        v16bf af = load_rm_frag(X,    HH, mt * 16, kt);
        v16bf bf = load_rm_frag(Bsrc, HH, brow,    kt);
        acc = wmma_bf16(af, bf, acc);
    }
    int lane = threadIdx.x & 31;
    int col  = nt * 16 + (lane & 15);
    float bias = bih[col] + bhh[col];
#pragma unroll
    for (int i = 0; i < 8; ++i) {
        int row = mt * 16 + i + ((lane >> 4) ? 8 : 0);
        G[(size_t)row * 1024 + col] = f2b(acc[i] + bias);
    }
}

// Persistent LSTM: one WGP workgroup, 32 waves, 512 sequential steps.
// h (bf16), c (f32), gate pre-activations (bf16) live in LDS (112 KB < 320 KB/WGP).
__global__ void lstm_kernel(const bf16_t* __restrict__ G, const bf16_t* __restrict__ Whhb,
                            const int* __restrict__ len, float* __restrict__ sel) {
    extern __shared__ char smem[];
    bf16_t* hS = (bf16_t*)smem;                                // 32 x 256 bf16
    float*  cS = (float*)(smem + HB * HH * 2);                 // 32 x 256 f32
    bf16_t* gS = (bf16_t*)(smem + HB * HH * 2 + HB * HH * 4);  // 32 x 1024 bf16
    int tid = threadIdx.x, lane = tid & 31, w = tid >> 5;
    for (int i = tid; i < HB * HH; i += 1024) { hS[i] = f2b(0.f); cS[i] = 0.f; }
    __syncthreads();
    for (int t = 0; t < TT; ++t) {
        const bf16_t* Gt = G + (size_t)t * HB * 1024;
#pragma unroll
        for (int q = 0; q < 4; ++q) {
            int tile = w + q * 32;            // 128 tiles: 2 Mtiles x 64 Ntiles
            int mt = tile & 1, nt = tile >> 1;
            int col = nt * 16 + (lane & 15);
            v8f acc;
#pragma unroll
            for (int i = 0; i < 8; ++i) {
                int row = mt * 16 + i + ((lane >> 4) ? 8 : 0);
                acc[i] = b2f(Gt[(size_t)row * 1024 + col]);
            }
#pragma unroll
            for (int kt = 0; kt < HH; kt += 32) {
                __builtin_prefetch(Whhb + (size_t)(nt * 16) * HH + kt + 32, 0, 0);
                v16bf af = load_rm_frag(hS,   HH, mt * 16, kt);
                v16bf bf = load_rm_frag(Whhb, HH, nt * 16, kt);
                acc = wmma_bf16(af, bf, acc);
            }
#pragma unroll
            for (int i = 0; i < 8; ++i) {
                int row = mt * 16 + i + ((lane >> 4) ? 8 : 0);
                gS[row * 1024 + col] = f2b(acc[i]);
            }
        }
        __syncthreads();
#pragma unroll
        for (int q = 0; q < 8; ++q) {
            int idx = tid + q * 1024;         // 8192 (b, j) cells
            int b = idx >> 8, j = idx & 255;
            float gi = b2f(gS[b * 1024 + j]);
            float gf = b2f(gS[b * 1024 + 256 + j]);
            float gg = b2f(gS[b * 1024 + 512 + j]);
            float go = b2f(gS[b * 1024 + 768 + j]);
            float cn = sigf(gf) * cS[idx] + sigf(gi) * tanhf(gg);
            float hn = sigf(go) * tanhf(cn);
            cS[idx] = cn;
            hS[idx] = f2b(hn);
            if (t == len[b] - 1) sel[idx] = hn;   // out[t] == h_t
        }
        __syncthreads();
    }
}

// Residual gate on selected rows only: fa = o + sig(o@W3+b3)*leaky(sig(o@W1+b1)*leaky(o@W2+b2))
__global__ void gate_kernel(const float* __restrict__ o,
                            const bf16_t* __restrict__ W1b, const float* __restrict__ b1,
                            const bf16_t* __restrict__ W2b, const float* __restrict__ b2,
                            const bf16_t* __restrict__ W3b, const float* __restrict__ b3,
                            float* __restrict__ fa) {
    __shared__ bf16_t oS[HB * HH];
    int lane = threadIdx.x & 31, w = threadIdx.x >> 5;
    for (int i = threadIdx.x; i < HB * HH; i += 512) oS[i] = f2b(o[i]);
    __syncthreads();
#pragma unroll
    for (int q = 0; q < 2; ++q) {
        int tile = w + q * 16;                // 32 tiles: 2 Mtiles x 16 Ntiles
        int mt = tile & 1, nt = tile >> 1;
        v8f a1 = {}, a2 = {}, a3 = {};
#pragma unroll
        for (int kt = 0; kt < HH; kt += 32) {
            v16bf af = load_rm_frag(oS, HH, mt * 16, kt);
            a1 = wmma_bf16(af, load_rm_frag(W1b, HH, nt * 16, kt), a1);
            a2 = wmma_bf16(af, load_rm_frag(W2b, HH, nt * 16, kt), a2);
            a3 = wmma_bf16(af, load_rm_frag(W3b, HH, nt * 16, kt), a3);
        }
        int col = nt * 16 + (lane & 15);
#pragma unroll
        for (int i = 0; i < 8; ++i) {
            int row = mt * 16 + i + ((lane >> 4) ? 8 : 0);
            float ov = o[row * HH + col];
            float c  = sigf(a1[i] + b1[col]) * leakyf(a2[i] + b2[col]);
            fa[row * HH + col] = ov + sigf(a3[i] + b3[col]) * leakyf(c);
        }
    }
}

// loss[b] = exp(||fa[b]-fb[b]+1e-6||)
__global__ void loss_kernel(const float* __restrict__ fa, const float* __restrict__ fb,
                            float* __restrict__ out) {
    int b = threadIdx.x;   // 32 lanes
    float s = 0.f;
    for (int j = 0; j < HH; ++j) {
        float d = fa[b * HH + j] - fb[b * HH + j] + 1e-6f;
        s += d * d;
    }
    out[b] = __expf(sqrtf(s));
}

// ---------------- host orchestration ----------------
extern "C" void kernel_launch(void* const* d_in, const int* in_sizes, int n_in,
                              void* d_out, int out_size, void* d_ws, size_t ws_size,
                              hipStream_t stream) {
    const float* anchor   = (const float*)d_in[0];
    const float* trajs    = (const float*)d_in[1];
    const float* negative = (const float*)d_in[2];
    const int*   alen     = (const int*)d_in[3];
    const int*   tlen     = (const int*)d_in[4];
    const int*   nlen     = (const int*)d_in[5];
    const float* Wm  = (const float*)d_in[6];
    const float* bm  = (const float*)d_in[7];
    const float* Wih = (const float*)d_in[8];
    const float* Whh = (const float*)d_in[9];
    const float* bih = (const float*)d_in[10];
    const float* bhh = (const float*)d_in[11];
    const float* W1  = (const float*)d_in[12];
    const float* b1  = (const float*)d_in[13];
    const float* W2  = (const float*)d_in[14];
    const float* b2  = (const float*)d_in[15];
    const float* W3  = (const float*)d_in[16];
    const float* b3  = (const float*)d_in[17];
    float* out = (float*)d_out;

    // workspace partition (all chunks 256B-aligned sizes)
    char* p = (char*)d_ws;
    auto alloc = [&](size_t bytes) { char* r = p; p += (bytes + 255) & ~(size_t)255; return r; };
    bf16_t* embA = (bf16_t*)alloc((size_t)HB * TT * HHE * 2);
    bf16_t* embT = (bf16_t*)alloc((size_t)HB * TT * HHE * 2);
    bf16_t* embN = (bf16_t*)alloc((size_t)HB * TT * HHE * 2);
    int* maskA = (int*)alloc((size_t)HB * TT * 4);
    int* maskT = (int*)alloc((size_t)HB * TT * 4);
    int* maskN = (int*)alloc((size_t)HB * TT * 4);
    bf16_t* Sbuf = (bf16_t*)alloc((size_t)HB * TT * TT * 2);
    bf16_t* Pbuf = (bf16_t*)alloc((size_t)HB * TT * TT * 2);
    bf16_t* sumb = (bf16_t*)alloc((size_t)HB * TT * HHE * 2);
    bf16_t* Xbuf = (bf16_t*)alloc((size_t)TT * HB * HH * 2);
    bf16_t* Gbuf = (bf16_t*)alloc((size_t)TT * HB * 1024 * 2);
    bf16_t* Wihb = (bf16_t*)alloc((size_t)1024 * HH * 2);
    bf16_t* Whhb = (bf16_t*)alloc((size_t)1024 * HH * 2);
    bf16_t* W1b  = (bf16_t*)alloc((size_t)HH * HH * 2);
    bf16_t* W2b  = (bf16_t*)alloc((size_t)HH * HH * 2);
    bf16_t* W3b  = (bf16_t*)alloc((size_t)HH * HH * 2);
    float* oasel = (float*)alloc((size_t)HB * HH * 4);
    float* fabuf = (float*)alloc((size_t)HB * HH * 4);
    float* fbsel = (float*)alloc((size_t)HB * HH * 4);

    // weight conversion to bf16 (once per call; deterministic)
    cvt_kernel<<<(1024 * HH + 255) / 256, 256, 0, stream>>>(Wih, Wihb, 1024 * HH);
    cvt_kernel<<<(1024 * HH + 255) / 256, 256, 0, stream>>>(Whh, Whhb, 1024 * HH);
    cvt_kernel<<<(HH * HH + 255) / 256, 256, 0, stream>>>(W1, W1b, HH * HH);
    cvt_kernel<<<(HH * HH + 255) / 256, 256, 0, stream>>>(W2, W2b, HH * HH);
    cvt_kernel<<<(HH * HH + 255) / 256, 256, 0, stream>>>(W3, W3b, HH * HH);

    // embeddings + masks
    embed_kernel<<<HB * TT, 128, 0, stream>>>(anchor,   Wm, bm, embA, maskA);
    embed_kernel<<<HB * TT, 128, 0, stream>>>(trajs,    Wm, bm, embT, maskT);
    embed_kernel<<<HB * TT, 128, 0, stream>>>(negative, Wm, bm, embN, maskN);

    const int LSTM_SMEM  = HB * HH * 2 + HB * HH * 4 + HB * 1024 * 2; // 112 KB
    const int GGEMM_SMEM = HAVE_TDM ? (8 * 16 * 256 * 2) : 0;         // 64 KB staging

    auto direction = [&](const bf16_t* eA, const int* mA, const bf16_t* eB, const int* mB,
                         const int* len, float* sel) {
        scores_kernel <<<4096,  256, 0, stream>>>(eA, eB, mA, mB, Sbuf);
        softmax_kernel<<<HB * TT, 256, 0, stream>>>(Sbuf, mA, mB, Pbuf);
        attnsum_kernel<<<1024,  256, 0, stream>>>(Pbuf, eB, sumb);
        pack_kernel   <<<HB * TT, 128, 0, stream>>>(eA, sumb, Xbuf);
        ggemm_kernel  <<<8192,  256, GGEMM_SMEM, stream>>>(Xbuf, Wihb, bih, bhh, Gbuf);
        lstm_kernel   <<<1, 1024, LSTM_SMEM, stream>>>(Gbuf, Whhb, len, sel);
    };

    // pair 1: anchor vs trajs
    direction(embA, maskA, embT, maskT, alen, oasel);
    gate_kernel<<<1, 512, 0, stream>>>(oasel, W1b, b1, W2b, b2, W3b, b3, fabuf);
    direction(embT, maskT, embA, maskA, tlen, fbsel);          // fb = raw LSTM output
    loss_kernel<<<1, 32, 0, stream>>>(fabuf, fbsel, out + 0);

    // pair 2: anchor vs negative
    direction(embA, maskA, embN, maskN, alen, oasel);
    gate_kernel<<<1, 512, 0, stream>>>(oasel, W1b, b1, W2b, b2, W3b, b3, fabuf);
    direction(embN, maskN, embA, maskA, nlen, fbsel);
    loss_kernel<<<1, 32, 0, stream>>>(fabuf, fbsel, out + 32);
}
